// PointNetMSG_47579647705389
// MI455X (gfx1250) — matrix operand order repaired and checked by reading, not compiled, not run on hardware
//
#include <hip/hip_runtime.h>
#include <math.h>

typedef __attribute__((ext_vector_type(16))) _Float16 v16h;
typedef __attribute__((ext_vector_type(8)))  _Float16 v8h;
typedef __attribute__((ext_vector_type(8)))  float    v8f;

#define KNB 64
#define NEG_INF (-3.402823466e38f)

// ---------------------------------------------------------------------------
// Fused GEMM on the WMMA pipe:  C[M,N] = act( scale * (A[M,K] @ W[K,N] + b) )
// A: row-major f32 (lda, rows 16B-aligned by construction).
// Wt: pre-transposed/pre-converted weights, f16 [Npad64][Kpad32], zero-padded,
//     so B staging is a single unconditional b128 copy global->LDS.
// Block = 256 threads = 8 wave32s; 64x64 tile; K stepped by 32; LDS
// double-buffered (one barrier per k-step). Main k-loop is branch-free; the
// guarded path runs only for the final partial chunk or row-edge tiles.
// ---------------------------------------------------------------------------
__global__ __launch_bounds__(256)
void gemm_wmma_kernel(const float* __restrict__ A, int lda,
                      const _Float16* __restrict__ Wt, int ldwt,
                      const float* __restrict__ bias,
                      int K, int N, float* __restrict__ C, int ldc,
                      int M, int doRelu, float scale)
{
    __shared__ _Float16 As[2][64][40];   // [buf][row][k]  (80B rows, 16B aligned)
    __shared__ _Float16 Bs[2][64][40];   // [buf][n][k]

    const int t       = threadIdx.x;
    const int rowBase = blockIdx.x * 64;
    const int colBase = blockIdx.y * 64;

    // cooperative-load coordinates: 64 rows x 4 groups of 8 k-columns
    const int lrow = t >> 2;
    const int cg   = (t & 3) * 8;

    // wave/lane coordinates
    const int lane = t & 31;
    const int wv   = t >> 5;
    const int rt   = wv & 3;       // 16-row tile within block
    const int ch   = wv >> 2;      // 32-col half within block
    const int mloc = lane & 15;
    const int hi   = lane >> 4;

    v8f acc0 = {0.f,0.f,0.f,0.f,0.f,0.f,0.f,0.f};
    v8f acc1 = {0.f,0.f,0.f,0.f,0.f,0.f,0.f,0.f};

    // shared compute step: fragments from LDS buffer pbuf, 2 WMMAs
    auto compute = [&](int pbuf) {
        const _Float16* arow = &As[pbuf][rt * 16 + mloc][0];
        v8h a_lo = *(const v8h*)(arow + hi * 16);
        v8h a_hi = *(const v8h*)(arow + hi * 16 + 8);
        v16h afrag = __builtin_shufflevector(a_lo, a_hi,
                        0,1,2,3,4,5,6,7,8,9,10,11,12,13,14,15);
        {
            const _Float16* brow = &Bs[pbuf][ch * 32 + mloc][0];
            v8h b_lo = *(const v8h*)(brow + hi * 16);
            v8h b_hi = *(const v8h*)(brow + hi * 16 + 8);
            v16h bfrag = __builtin_shufflevector(b_lo, b_hi,
                            0,1,2,3,4,5,6,7,8,9,10,11,12,13,14,15);
            acc0 = __builtin_amdgcn_wmma_f32_16x16x32_f16(
                       false, afrag, false, bfrag, (short)0, acc0, false, false);
        }
        {
            const _Float16* brow = &Bs[pbuf][ch * 32 + 16 + mloc][0];
            v8h b_lo = *(const v8h*)(brow + hi * 16);
            v8h b_hi = *(const v8h*)(brow + hi * 16 + 8);
            v16h bfrag = __builtin_shufflevector(b_lo, b_hi,
                            0,1,2,3,4,5,6,7,8,9,10,11,12,13,14,15);
            acc1 = __builtin_amdgcn_wmma_f32_16x16x32_f16(
                       false, afrag, false, bfrag, (short)0, acc1, false, false);
        }
    };

    const bool rowFull = (rowBase + 64) <= M;
    const int  kMain   = rowFull ? (K & ~31) : 0;   // full chunks, fast path

    const _Float16* wpBase = Wt + (size_t)(colBase + lrow) * ldwt + cg;
    const float*    apBase = A  + (size_t)(rowBase + lrow) * lda + cg;

    int pb = 0, k0 = 0;
    // ---- main loop: branch-free; all loads issue before LDS stores ----
    for (; k0 < kMain; k0 += 32, pb ^= 1) {
        const v8h    breg = *(const v8h*)(wpBase + k0);
        const float* ap   = apBase + k0;
        const float4 a0   = *(const float4*)ap;
        const float4 a1   = *(const float4*)(ap + 4);
        __builtin_prefetch(ap + 32, 0, 1);   // next k-chunk of this row
        v8h h;
        h[0]=(_Float16)a0.x; h[1]=(_Float16)a0.y;
        h[2]=(_Float16)a0.z; h[3]=(_Float16)a0.w;
        h[4]=(_Float16)a1.x; h[5]=(_Float16)a1.y;
        h[6]=(_Float16)a1.z; h[7]=(_Float16)a1.w;
        *(v8h*)&Bs[pb][lrow][cg] = breg;
        *(v8h*)&As[pb][lrow][cg] = h;
        __syncthreads();                     // 1 barrier/step: safe w/ 2 buffers
        compute(pb);
    }
    // ---- tail: partial k-chunk and/or row-edge tiles (guards on loads) ----
    for (; k0 < K; k0 += 32, pb ^= 1) {
        const v8h breg = *(const v8h*)(wpBase + k0);
        const int gr = rowBase + lrow;
        v8h h;
        #pragma unroll
        for (int j = 0; j < 8; ++j) {
            const int kk = k0 + cg + j;
            float v = 0.f;
            if (gr < M && kk < K) v = A[(size_t)gr * lda + kk];
            h[j] = (_Float16)v;
        }
        *(v8h*)&Bs[pb][lrow][cg] = breg;
        *(v8h*)&As[pb][lrow][cg] = h;
        __syncthreads();
        compute(pb);
    }

    // ---- epilogue: 16x16 f32 C layout (VGPR v -> M = v + hi*8, N = lane&15) ----
    auto store_tile = [&](const v8f& acc, int j2) {
        const int gn = colBase + ch * 32 + j2 * 16 + mloc;
        if (gn < N) {
            const float bv = bias ? bias[gn] : 0.f;
            #pragma unroll
            for (int v = 0; v < 8; ++v) {
                const int gm = rowBase + rt * 16 + hi * 8 + v;
                if (gm < M) {
                    float val = (acc[v] + bv) * scale;
                    if (doRelu) val = fmaxf(val, 0.f);
                    C[(size_t)gm * ldc + gn] = val;
                }
            }
        }
    };
    store_tile(acc0, 0);
    store_tile(acc1, 1);
}

// ---------------------------------------------------------------------------
// Wt[n*Kpad + k] = f16( W[k*N + n] ), zero-padded to [Npad64, Kpad32]
// ---------------------------------------------------------------------------
__global__ void convert_w_kernel(const float* __restrict__ W, int K, int N,
                                 _Float16* __restrict__ Wt, int Kpad, long total)
{
    long g = (long)blockIdx.x * blockDim.x + threadIdx.x;
    if (g >= total) return;
    const int n = (int)(g / Kpad);
    const int k = (int)(g % Kpad);
    float v = (n < N && k < K) ? W[(size_t)k * N + n] : 0.f;
    Wt[g] = (_Float16)v;
}

// ---------------------------------------------------------------------------
// out[r*ldo + c] = cat(A[r,ca], B[r,cb])[c]
// ---------------------------------------------------------------------------
__global__ void concat2_kernel(const float* __restrict__ A, int ca,
                               const float* __restrict__ B, int cb,
                               long total, int ldo, float* __restrict__ out)
{
    long g = (long)blockIdx.x * blockDim.x + threadIdx.x;
    if (g >= total) return;
    const int cols = ca + cb;
    long r = g / cols; int c = (int)(g % cols);
    out[r * (long)ldo + c] = (c < ca) ? A[r * (long)ca + c]
                                      : B[r * (long)cb + (c - ca)];
}

// ---------------------------------------------------------------------------
// out[c] = max over rows of H[i*ldh + c]   (one block per column)
// ---------------------------------------------------------------------------
__global__ __launch_bounds__(256)
void colmax_kernel(const float* __restrict__ H, int M, int ldh,
                   float* __restrict__ out)
{
    const int c = blockIdx.x;
    __shared__ float red[256];
    float best = NEG_INF;
    for (int i = threadIdx.x; i < M; i += 256)
        best = fmaxf(best, H[(size_t)i * ldh + c]);
    red[threadIdx.x] = best;
    __syncthreads();
    for (int off = 128; off > 0; off >>= 1) {
        if (threadIdx.x < off)
            red[threadIdx.x] = fmaxf(red[threadIdx.x], red[threadIdx.x + off]);
        __syncthreads();
    }
    if (threadIdx.x == 0) out[c] = red[0];
}

// ---------------------------------------------------------------------------
// pos_t[b,n] = pos[b,n] @ (T_b.reshape(3,3) + I)
// ---------------------------------------------------------------------------
__global__ void transform_kernel(const float* __restrict__ pos,
                                 const float* __restrict__ t9,
                                 int N, int B, float* __restrict__ out)
{
    int g = blockIdx.x * blockDim.x + threadIdx.x;
    if (g >= B * N) return;
    const int b = g / N;
    const float* T = t9 + b * 9;
    const float* p = pos + (size_t)g * 3;
    float p0 = p[0], p1 = p[1], p2 = p[2];
    float* o = out + (size_t)g * 3;
    o[0] = p0 * (T[0] + 1.f) + p1 * T[3] + p2 * T[6];
    o[1] = p0 * T[1] + p1 * (T[4] + 1.f) + p2 * T[7];
    o[2] = p0 * T[2] + p1 * T[5] + p2 * (T[8] + 1.f);
}

// ---------------------------------------------------------------------------
// Farthest point sampling: one block per batch; mind[] lives in LDS.
// Tie-break matches jnp.argmax (first occurrence of the max).
// ---------------------------------------------------------------------------
__global__ __launch_bounds__(256)
void fps_kernel(const float* __restrict__ pos, int N, int M,
                int* __restrict__ idx_out)
{
    pos     += (size_t)blockIdx.x * N * 3;
    idx_out += (size_t)blockIdx.x * M;

    __shared__ float mind[4096];
    __shared__ float redv[256];
    __shared__ int   redi[256];
    __shared__ float lastp[3];

    const int t = threadIdx.x;
    for (int i = t; i < N; i += 256) mind[i] = 3.4e38f;
    if (t == 0) {
        idx_out[0] = 0;
        lastp[0] = pos[0]; lastp[1] = pos[1]; lastp[2] = pos[2];
    }
    __syncthreads();

    for (int s = 1; s < M; ++s) {
        const float px = lastp[0], py = lastp[1], pz = lastp[2];
        float bestv = NEG_INF; int besti = N;
        for (int i = t; i < N; i += 256) {
            const float dx = pos[i*3+0] - px;
            const float dy = pos[i*3+1] - py;
            const float dz = pos[i*3+2] - pz;
            float mv = fminf(mind[i], dx*dx + dy*dy + dz*dz);
            mind[i] = mv;
            if (mv > bestv) { bestv = mv; besti = i; }
        }
        redv[t] = bestv; redi[t] = besti;
        __syncthreads();
        for (int off = 128; off > 0; off >>= 1) {
            if (t < off) {
                const float ov = redv[t + off]; const int oi = redi[t + off];
                if (ov > redv[t] || (ov == redv[t] && oi < redi[t])) {
                    redv[t] = ov; redi[t] = oi;
                }
            }
            __syncthreads();
        }
        if (t == 0) {
            const int nx = redi[0];
            idx_out[s] = nx;
            lastp[0] = pos[nx*3+0]; lastp[1] = pos[nx*3+1]; lastp[2] = pos[nx*3+2];
        }
        __syncthreads();
    }
}

// ---------------------------------------------------------------------------
// out[b,m,:] = pos[b, idx[b,m], :]
// ---------------------------------------------------------------------------
__global__ void gather_pos_kernel(const float* __restrict__ pos, int posStride,
                                  const int* __restrict__ idx, int M,
                                  float* __restrict__ out, int total)
{
    int g = blockIdx.x * blockDim.x + threadIdx.x;
    if (g >= total) return;
    const int b = g / (M * 3);
    const int rem = g % (M * 3);
    const int m = rem / 3, d = rem % 3;
    out[g] = pos[(size_t)b * posStride + (size_t)idx[b * M + m] * 3 + d];
}

// ---------------------------------------------------------------------------
// Ball query: first K=64 in-radius source indices, in index order (stable),
// matching argsort(where(mask,0,1))[:, :K] of the reference.
// ---------------------------------------------------------------------------
__global__ void ball_query_kernel(const float* __restrict__ pos_src, int Nsrc,
                                  const float* __restrict__ pos_q, int Mq,
                                  float r2, int* __restrict__ idxK,
                                  int* __restrict__ valK)
{
    int m = blockIdx.x * blockDim.x + threadIdx.x;
    if (m >= Mq) return;
    const float qx = pos_q[m*3+0], qy = pos_q[m*3+1], qz = pos_q[m*3+2];
    int cnt = 0;
    for (int i = 0; i < Nsrc && cnt < KNB; ++i) {
        const float dx = pos_src[i*3+0] - qx;
        const float dy = pos_src[i*3+1] - qy;
        const float dz = pos_src[i*3+2] - qz;
        if (dx*dx + dy*dy + dz*dz <= r2) {
            idxK[m*KNB + cnt] = i;
            valK[m*KNB + cnt] = 1;
            ++cnt;
        }
    }
    for (; cnt < KNB; ++cnt) { idxK[m*KNB + cnt] = 0; valK[m*KNB + cnt] = 0; }
}

// ---------------------------------------------------------------------------
// Edge features: E[(m*K+k)*ldE + c] = cat(x_src[j], pos_src[j]-pos_q[m])[c]
// Invalid edges are zero-filled (masked out in the max-pool).
// ---------------------------------------------------------------------------
__global__ void edge_feat_kernel(const float* __restrict__ x_src, int C,
                                 const float* __restrict__ pos_src,
                                 const float* __restrict__ pos_q,
                                 const int* __restrict__ idxK,
                                 const int* __restrict__ valK,
                                 long total, int ldE, float* __restrict__ E)
{
    long g = (long)blockIdx.x * blockDim.x + threadIdx.x;
    if (g >= total) return;
    const int cols = C + 3;
    const long e = g / cols;
    const int c = (int)(g % cols);
    const int m = (int)(e >> 6);  // K == 64
    float v = 0.f;
    if (valK[e]) {
        const int j = idxK[e];
        if (c < C) v = x_src[(size_t)j * C + c];
        else { const int d = c - C; v = pos_src[j*3 + d] - pos_q[m*3 + d]; }
    }
    E[e * (long)ldE + c] = v;
}

// ---------------------------------------------------------------------------
// Masked max over K neighbors; 0 if no valid neighbor.
// ---------------------------------------------------------------------------
__global__ void edge_maxpool_kernel(const float* __restrict__ H, int ldh,
                                    const int* __restrict__ valK,
                                    int Mq, int Cout,
                                    float* __restrict__ out, int ldo)
{
    int g = blockIdx.x * blockDim.x + threadIdx.x;
    if (g >= Mq * Cout) return;
    const int m = g / Cout, c = g % Cout;
    const float* hp = H + (size_t)m * KNB * ldh + c;
    const int* vp = valK + m * KNB;
    float best = NEG_INF; int any = 0;
    for (int k = 0; k < KNB; ++k) {
        if (vp[k]) { any = 1; best = fmaxf(best, hp[(size_t)k * ldh]); }
    }
    out[(size_t)m * ldo + c] = any ? best : 0.f;
}

// ===========================================================================
extern "C" void kernel_launch(void* const* d_in, const int* in_sizes, int n_in,
                              void* d_out, int out_size, void* d_ws, size_t ws_size,
                              hipStream_t stream)
{
    (void)out_size; (void)ws_size;
    const int B = 4, N = 4096, M1 = 1024, M2 = 512;
    const float BN_SCALE = 1.0f / sqrtf(1.0f + 1e-5f);

    // ---- locate pos / x / params regardless of pytree flattening order ----
    int posIdx, xIdx, pbase;
    if (in_sizes[0] == B * N * 3) { posIdx = 0; xIdx = 1; pbase = 2; }
    else                          { pbase = 0; posIdx = n_in - 2; xIdx = n_in - 1; }
    const float* pos = (const float*)d_in[posIdx];
    const float* x   = (const float*)d_in[xIdx];

    int oHead, oSa1, oSa2, oSa3, oT1, oT2;
    if (in_sizes[pbase] == 1024 * 512) {  // sorted keys: head, sa1, sa2, sa3, tnet1, tnet2
        oHead = pbase + 0;  oSa1 = pbase + 6;  oSa2 = pbase + 24;
        oSa3  = pbase + 42; oT1  = pbase + 48; oT2  = pbase + 54;
    } else {                              // insertion order: tnet1, tnet2, sa1, sa2, sa3, head
        oT1  = pbase + 0;  oT2  = pbase + 6;  oSa1  = pbase + 12;
        oSa2 = pbase + 30; oSa3 = pbase + 48; oHead = pbase + 54;
    }

    static const int t1dims[4] = {4, 64, 128, 1024};
    static const int t2dims[4] = {1024, 512, 256, 9};
    static const int s1dims[3][4] = {{4,32,32,64},{4,64,64,128},{4,64,96,128}};
    static const int s2dims[3][4] = {{323,64,64,128},{323,128,128,256},{323,128,128,256}};
    static const int s3dims[4] = {643, 256, 512, 1024};
    static const int hdims[4]  = {1024, 512, 256, 1};

    // ---- carve workspace ----
    char* ws = (char*)d_ws;
    size_t off = 0;
    auto carve = [&](size_t bytes) -> char* {
        char* p = ws + off;
        off = (off + bytes + 255) & ~(size_t)255;
        return p;
    };
    float* POS_T = (float*)carve((size_t)B * N * 3 * 4);
    float* G1    = (float*)carve((size_t)B * 1024 * 4);
    float* T9    = (float*)carve((size_t)B * 9 * 4);
    int*   FPS1  = (int*)  carve((size_t)B * M1 * 4);
    float* POS1  = (float*)carve((size_t)B * M1 * 3 * 4);
    float* X1    = (float*)carve((size_t)B * M1 * 320 * 4);
    int*   FPS2  = (int*)  carve((size_t)B * M2 * 4);
    float* POS2  = (float*)carve((size_t)B * M2 * 3 * 4);
    float* X2    = (float*)carve((size_t)B * M2 * 640 * 4);
    int*   IDXK  = (int*)  carve((size_t)M1 * KNB * 4);
    int*   VALK  = (int*)  carve((size_t)M1 * KNB * 4);
    float* XIN3  = (float*)carve((size_t)B * M2 * 644 * 4);   // ld 644 (padded)
    float* G3    = (float*)carve((size_t)B * 1024 * 4);
    float* BUF_A = (float*)carve((size_t)48 * 1024 * 1024);
    float* BUF_B = (float*)carve((size_t)48 * 1024 * 1024);

    // ---- pre-transpose + f16-convert all weights (zero-padded) ----
    struct Lyr { const float* b; _Float16* Wt; int K, N, Kpad; };
    Lyr Ls[30]; int nl = 0;
    auto addMlp3 = [&](int gbase, const int* dims) {
        for (int i = 0; i < 3; ++i) {
            Lyr& l = Ls[nl++];
            const float* W = (const float*)d_in[gbase + 2*i];
            l.b = (const float*)d_in[gbase + 2*i + 1];
            l.K = dims[i]; l.N = dims[i+1];
            l.Kpad = (l.K + 31) & ~31;
            const int Npad = (l.N + 63) & ~63;
            l.Wt = (_Float16*)carve((size_t)Npad * l.Kpad * 2);
            long tot = (long)Npad * l.Kpad;
            convert_w_kernel<<<(unsigned)((tot + 255) / 256), 256, 0, stream>>>(
                W, l.K, l.N, l.Wt, l.Kpad, tot);
        }
    };
    addMlp3(oT1, t1dims);                               // layers 0..2
    addMlp3(oT2, t2dims);                               // layers 3..5
    for (int s = 0; s < 3; ++s) addMlp3(oSa1 + s*6, s1dims[s]);  // 6..14
    for (int s = 0; s < 3; ++s) addMlp3(oSa2 + s*6, s2dims[s]);  // 15..23
    addMlp3(oSa3, s3dims);                              // 24..26
    addMlp3(oHead, hdims);                              // 27..29

    auto gemm = [&](const float* A, int lda, const Lyr& l,
                    float* C, int ldc, int M, int relu, float scale) {
        dim3 grid((M + 63) / 64, (l.N + 63) / 64);
        gemm_wmma_kernel<<<grid, 256, 0, stream>>>(A, lda, l.Wt, l.Kpad, l.b,
                                                   l.K, l.N, C, ldc, M, relu, scale);
    };
    // 3-layer MLP (plain_last): L0 A->BUF_B, L1 BUF_B->BUF_A, L2 BUF_A->dest.
    auto mlp3 = [&](const float* A, int lda, int M, int lbase,
                    float* dest, int ldd, int useBN) {
        const Lyr* l = &Ls[lbase];
        const float s = useBN ? BN_SCALE : 1.f;
        gemm(A,     lda,    l[0], BUF_B, l[0].N, M, 1, s);
        gemm(BUF_B, l[0].N, l[1], BUF_A, l[1].N, M, 1, s);
        gemm(BUF_A, l[1].N, l[2], dest, ldd, M, 0, 1.f);
    };

    // ================= TNet: per-point MLP + global max (per batch) =========
    for (int b = 0; b < B; ++b) {
        const float* posb = pos + (size_t)b * N * 3;
        const float* xb   = x   + (size_t)b * N;
        long tot = (long)N * 4;
        concat2_kernel<<<(unsigned)((tot + 255) / 256), 256, 0, stream>>>(
            posb, 3, xb, 1, tot, 4, BUF_A);
        mlp3(BUF_A, 4, N, 0, BUF_B, 1024, 1);
        colmax_kernel<<<1024, 256, 0, stream>>>(BUF_B, N, 1024, G1 + b * 1024);
    }
    mlp3(G1, 1024, B, 3, T9, 9, 1);    // TNet second MLP -> T9[4,9]
    {
        int tot = B * N;
        transform_kernel<<<(tot + 255) / 256, 256, 0, stream>>>(pos, T9, N, B, POS_T);
    }

    // ================= SA1 (MSG on pos_t, x; M=1024, K=64) ==================
    fps_kernel<<<B, 256, 0, stream>>>(POS_T, N, M1, FPS1);
    {
        int tot = B * M1 * 3;
        gather_pos_kernel<<<(tot + 255) / 256, 256, 0, stream>>>(
            POS_T, N * 3, FPS1, M1, POS1, tot);
    }
    {
        const float r2_1[3] = {1.f, 4.f, 16.f};
        const int   off1[3] = {0, 64, 192};
        for (int b = 0; b < B; ++b) {
            const float* ps = POS_T + (size_t)b * N * 3;
            const float* pq = POS1  + (size_t)b * M1 * 3;
            const float* xs = x     + (size_t)b * N;       // C = 1
            for (int s = 0; s < 3; ++s) {
                ball_query_kernel<<<(M1 + 255) / 256, 256, 0, stream>>>(
                    ps, N, pq, M1, r2_1[s], IDXK, VALK);
                long tot = (long)M1 * KNB * 4;             // C+3 = 4 cols (ld 4)
                edge_feat_kernel<<<(unsigned)((tot + 255) / 256), 256, 0, stream>>>(
                    xs, 1, ps, pq, IDXK, VALK, tot, 4, BUF_A);
                mlp3(BUF_A, 4, M1 * KNB, 6 + s * 3, BUF_B, s1dims[s][3], 1);
                int tot2 = M1 * s1dims[s][3];
                edge_maxpool_kernel<<<(tot2 + 255) / 256, 256, 0, stream>>>(
                    BUF_B, s1dims[s][3], VALK, M1, s1dims[s][3],
                    X1 + (size_t)b * M1 * 320 + off1[s], 320);
            }
        }
    }

    // ================= SA2 (MSG on pos1, x1; M=512, K=64) ===================
    fps_kernel<<<B, 256, 0, stream>>>(POS1, M1, M2, FPS2);
    {
        int tot = B * M2 * 3;
        gather_pos_kernel<<<(tot + 255) / 256, 256, 0, stream>>>(
            POS1, M1 * 3, FPS2, M2, POS2, tot);
    }
    {
        const float r2_2[3] = {4.f, 16.f, 64.f};
        const int   off2[3] = {0, 128, 384};
        for (int b = 0; b < B; ++b) {
            const float* ps = POS1 + (size_t)b * M1 * 3;
            const float* pq = POS2 + (size_t)b * M2 * 3;
            const float* xs = X1   + (size_t)b * M1 * 320; // C = 320
            for (int s = 0; s < 3; ++s) {
                ball_query_kernel<<<(M2 + 255) / 256, 256, 0, stream>>>(
                    ps, M1, pq, M2, r2_2[s], IDXK, VALK);
                long tot = (long)M2 * KNB * 323;           // logical cols, ld 324
                edge_feat_kernel<<<(unsigned)((tot + 255) / 256), 256, 0, stream>>>(
                    xs, 320, ps, pq, IDXK, VALK, tot, 324, BUF_A);
                mlp3(BUF_A, 324, M2 * KNB, 15 + s * 3, BUF_B, s2dims[s][3], 1);
                int tot2 = M2 * s2dims[s][3];
                edge_maxpool_kernel<<<(tot2 + 255) / 256, 256, 0, stream>>>(
                    BUF_B, s2dims[s][3], VALK, M2, s2dims[s][3],
                    X2 + (size_t)b * M2 * 640 + off2[s], 640);
            }
        }
    }

    // ================= SA3: global per-point MLP + max =======================
    {
        long tot = (long)B * M2 * 643;                     // logical cols, ld 644
        concat2_kernel<<<(unsigned)((tot + 255) / 256), 256, 0, stream>>>(
            X2, 640, POS2, 3, tot, 644, XIN3);
        mlp3(XIN3, 644, B * M2, 24, BUF_B, 1024, 1);
        for (int b = 0; b < B; ++b)
            colmax_kernel<<<1024, 256, 0, stream>>>(
                BUF_B + (size_t)b * M2 * 1024, M2, 1024, G3 + b * 1024);
    }

    // ================= Head MLP (no BN, ReLU on hidden) -> d_out [4,1] ======
    mlp3(G3, 1024, B, 27, (float*)d_out, 1, 0);
}
